// AFD_28784870818375
// MI455X (gfx1250) — compile-verified, matrix operand order checked by reading, not run on hardware
//
#include <hip/hip_runtime.h>
#include <math.h>

// ---------------- problem constants ----------------
#define BB   8
#define CC   64
#define DI   128
#define NS   16
#define RR   4
#define KD   4          // scan directions
#define HH   64
#define WW   64
#define LL   4096       // H*W
#define MTOT (BB*LL)    // 32768 pixels
#define SCH  64         // scan timesteps staged per LDS chunk
#define LN_EPS 1e-5f
#define CN_EPS 1e-6f

typedef __attribute__((ext_vector_type(16))) _Float16 v16h;
typedef __attribute__((ext_vector_type(8)))  float    v8f;
typedef unsigned int u32x4 __attribute__((ext_vector_type(4)));
typedef unsigned int u32x8 __attribute__((ext_vector_type(8)));

// ---------------- small math helpers ----------------
__device__ __forceinline__ float siluf(float v)     { return v / (1.f + __expf(-v)); }
__device__ __forceinline__ float softplusf(float v) { return v > 20.f ? v : log1pf(__expf(v)); }
__device__ __forceinline__ float geluf(float v)     { return 0.5f * v * (1.f + erff(v * 0.70710678118654752f)); }

// ---------------- WMMA fragment loaders (CDNA5 16x16x32 f16 layout) ----------------
// Row-major [rows, ld] fragment (A operand, or NT (out,in) weight as B operand).
// Lane half selects K octet: elems 0..7 -> K = kk+8*half+j, elems 8..15 -> K = kk+16+8*half+j.
// Contiguous in K -> vectorized float4 loads.
__device__ __forceinline__ v16h frag_rowmajor(const float* __restrict__ P, int ld, int m0, int kk) {
  int lane = threadIdx.x & 31, half = lane >> 4, r = lane & 15;
  const float* row = P + (size_t)(m0 + r) * ld + kk + 8 * half;
  float4 p0 = *(const float4*)(row);
  float4 p1 = *(const float4*)(row + 4);
  float4 q0 = *(const float4*)(row + 16);
  float4 q1 = *(const float4*)(row + 20);
  v16h a;
  a[0]  = (_Float16)p0.x; a[1]  = (_Float16)p0.y; a[2]  = (_Float16)p0.z; a[3]  = (_Float16)p0.w;
  a[4]  = (_Float16)p1.x; a[5]  = (_Float16)p1.y; a[6]  = (_Float16)p1.z; a[7]  = (_Float16)p1.w;
  a[8]  = (_Float16)q0.x; a[9]  = (_Float16)q0.y; a[10] = (_Float16)q0.z; a[11] = (_Float16)q0.w;
  a[12] = (_Float16)q1.x; a[13] = (_Float16)q1.y; a[14] = (_Float16)q1.z; a[15] = (_Float16)q1.w;
  return a;
}

// guarded version (rows >= Mv read as zero)
__device__ __forceinline__ v16h frag_rowmajor_guard(const float* __restrict__ P, int ld, int m0, int kk, int Mv) {
  int lane = threadIdx.x & 31, half = lane >> 4, r = lane & 15;
  int m = m0 + r;
  v16h a;
  if (m < Mv) {
    const float* row = P + (size_t)m * ld + kk + 8 * half;
    float4 p0 = *(const float4*)(row);
    float4 p1 = *(const float4*)(row + 4);
    float4 q0 = *(const float4*)(row + 16);
    float4 q1 = *(const float4*)(row + 20);
    a[0]  = (_Float16)p0.x; a[1]  = (_Float16)p0.y; a[2]  = (_Float16)p0.z; a[3]  = (_Float16)p0.w;
    a[4]  = (_Float16)p1.x; a[5]  = (_Float16)p1.y; a[6]  = (_Float16)p1.z; a[7]  = (_Float16)p1.w;
    a[8]  = (_Float16)q0.x; a[9]  = (_Float16)q0.y; a[10] = (_Float16)q0.z; a[11] = (_Float16)q0.w;
    a[12] = (_Float16)q1.x; a[13] = (_Float16)q1.y; a[14] = (_Float16)q1.z; a[15] = (_Float16)q1.w;
  } else {
#pragma unroll
    for (int j = 0; j < 16; ++j) a[j] = (_Float16)0.f;
  }
  return a;
}

// B fragment from a (K,N) row-major activation matrix: element (kk+K(j), n0+r)
__device__ __forceinline__ v16h frag_kn(const float* __restrict__ P, int ldb, int n0, int kk) {
  int lane = threadIdx.x & 31, half = lane >> 4, r = lane & 15;
  int n = n0 + r;
  v16h b;
#pragma unroll
  for (int j = 0; j < 8; ++j) { int k = kk + 8 * half + j;       b[j]     = (_Float16)P[(size_t)k * ldb + n]; }
#pragma unroll
  for (int j = 0; j < 8; ++j) { int k = kk + 16 + 8 * half + j;  b[8 + j] = (_Float16)P[(size_t)k * ldb + n]; }
  return b;
}

// A fragment gathered from an NCHW (B,Chan,L) tensor: logical A[m,k] = F[(batch*Chan+k)*L + l]
__device__ __forceinline__ v16h frag_nchw(const float* __restrict__ F, int Chan, int m0, int kk) {
  int lane = threadIdx.x & 31, half = lane >> 4, r = lane & 15;
  int m = m0 + r, b = m >> 12, l = m & (LL - 1);
  v16h a;
#pragma unroll
  for (int j = 0; j < 8; ++j) { int k = kk + 8 * half + j;       a[j]     = (_Float16)F[((size_t)(b * Chan + k)) * LL + l]; }
#pragma unroll
  for (int j = 0; j < 8; ++j) { int k = kk + 16 + 8 * half + j;  a[8 + j] = (_Float16)F[((size_t)(b * Chan + k)) * LL + l]; }
  return a;
}

#define WMMA_F16(a, b, c) __builtin_amdgcn_wmma_f32_16x16x32_f16(false, (a), false, (b), (short)0, (c), false, false)

// ================= kernels =================

// (1) LayerNorm over C per pixel, NCHW x -> NHWC xn
__global__ void k_ln1(const float* __restrict__ x, const float* __restrict__ g,
                      const float* __restrict__ be, float* __restrict__ xn) {
  int m = blockIdx.x * blockDim.x + threadIdx.x;
  if (m >= MTOT) return;
  int b = m >> 12, l = m & (LL - 1);
  const float* xb = x + (size_t)b * CC * LL + l;
  float s = 0.f, s2 = 0.f;
  for (int c = 0; c < CC; ++c) { float v = xb[(size_t)c * LL]; s += v; s2 += v * v; }
  float mu = s / CC, var = s2 / CC - mu * mu;
  float rs = rsqrtf(var + LN_EPS);
  float* o = xn + (size_t)m * CC;
  for (int c = 0; c < CC; ++c) o[c] = (xb[(size_t)c * LL] - mu) * rs * g[c] + be[c];
}

// (2) in_proj GEMM: xz[m,n] = sum_k xn[m,k]*W[n,k] + bias[n]  (M=32768,N=256,K=64)
// one wave: 16x64 of C (4 N-tiles), A fragment reused 4x
__global__ void k_inproj(const float* __restrict__ A, const float* __restrict__ W,
                         const float* __restrict__ bias, float* __restrict__ out) {
  int m0 = blockIdx.y * 16, nb = blockIdx.x * 64;
  v8f acc[4] = {};
#pragma unroll
  for (int kk = 0; kk < CC; kk += 32) {
    v16h a = frag_rowmajor(A, CC, m0, kk);
#pragma unroll
    for (int t = 0; t < 4; ++t) {
      v16h b = frag_rowmajor(W, CC, nb + 16 * t, kk);
      acc[t] = WMMA_F16(a, b, acc[t]);
    }
  }
  int lane = threadIdx.x & 31, half = lane >> 4, r = lane & 15;
#pragma unroll
  for (int t = 0; t < 4; ++t) {
    int n = nb + 16 * t + r;
    float bn = bias[n];
#pragma unroll
    for (int i = 0; i < 8; ++i) { int m = m0 + i + 8 * half; out[(size_t)m * 256 + n] = acc[t][i] + bn; }
  }
}

// (3) depthwise 3x3 conv + SiLU on xi (first 128 ch of xz, NHWC) -> xc (B,DI,L)
__global__ void k_conv_silu(const float* __restrict__ xz, const float* __restrict__ cw,
                            const float* __restrict__ cb, float* __restrict__ xc) {
  int t = blockIdx.x * blockDim.x + threadIdx.x;
  if (t >= BB * DI * LL) return;
  int b = t >> 19, rem = t & ((1 << 19) - 1);
  int d = rem >> 12, l = rem & (LL - 1);
  int h = l >> 6, w = l & 63;
  float acc = cb[d];
  for (int dy = 0; dy < 3; ++dy) {
    int y = h + dy - 1; if (y < 0 || y >= HH) continue;
    for (int dx = 0; dx < 3; ++dx) {
      int xw = w + dx - 1; if (xw < 0 || xw >= WW) continue;
      acc += cw[(d * 3 + dy) * 3 + dx] * xz[((size_t)(b << 12) + y * WW + xw) * 256 + d];
    }
  }
  xc[((size_t)(b * DI + d)) * LL + l] = siluf(acc);
}

// (4) build 4-direction xs from xc
__global__ void k_xs_build(const float* __restrict__ xc, float* __restrict__ xs) {
  int t = blockIdx.x * blockDim.x + threadIdx.x;
  if (t >= BB * KD * DI * LL) return;
  int l = t & (LL - 1), d = (t >> 12) & 127, k = (t >> 19) & 3, b = t >> 21;
  int ls;
  if (k == 0)      ls = l;
  else if (k == 1) ls = ((l & 63) << 6) | (l >> 6);
  else if (k == 2) ls = (LL - 1) - l;
  else { int l2 = (LL - 1) - l; ls = ((l2 & 63) << 6) | (l2 >> 6); }
  xs[t] = xc[((size_t)(b * DI + d)) * LL + ls];
}

// (5) x_dbl GEMM per (b,k): [36 x 128] x [128 x 4096]
// one wave: all 3 M-tiles (48x16 of C), activation B fragment reused 3x
__global__ void k_xdbl(const float* __restrict__ xproj, const float* __restrict__ xs,
                       float* __restrict__ xdbl) {
  int n0 = blockIdx.x * 16, bk = blockIdx.y;
  int k = bk & 3;
  const float* A  = xproj + (size_t)k * 36 * DI;
  const float* Bm = xs + (size_t)bk * DI * LL;
  v8f acc[3] = {};
#pragma unroll
  for (int kk = 0; kk < DI; kk += 32) {
    v16h b = frag_kn(Bm, LL, n0, kk);
#pragma unroll
    for (int t = 0; t < 3; ++t) {
      v16h a = frag_rowmajor_guard(A, DI, 16 * t, kk, 36);
      acc[t] = WMMA_F16(a, b, acc[t]);
    }
  }
  int lane = threadIdx.x & 31, half = lane >> 4, r = lane & 15;
  int n = n0 + r;
  float* Cp = xdbl + (size_t)bk * 36 * LL;
#pragma unroll
  for (int t = 0; t < 3; ++t)
#pragma unroll
    for (int i = 0; i < 8; ++i) {
      int m = 16 * t + i + 8 * half;
      if (m < 36) Cp[(size_t)m * LL + n] = acc[t][i];
    }
}

// (6) dt projection + softplus
__global__ void k_dt(const float* __restrict__ xdbl, const float* __restrict__ dtw,
                     const float* __restrict__ dtb, float* __restrict__ dts) {
  int t = blockIdx.x * blockDim.x + threadIdx.x;
  if (t >= BB * KD * DI * LL) return;
  int l = t & (LL - 1), d = (t >> 12) & 127, k = (t >> 19) & 3, b = t >> 21;
  const float* xd = xdbl + ((size_t)(b * KD + k)) * 36 * LL + l;
  float acc = dtb[k * DI + d];
#pragma unroll
  for (int r = 0; r < RR; ++r) acc += xd[(size_t)r * LL] * dtw[(k * DI + d) * RR + r];
  dts[t] = softplusf(acc);
}

// (7) selective scan: one block per (b,k), 128 threads (one per channel d).
// B_t/C_t rows (32 x SCH tile of x_dbl) staged into LDS per chunk via the
// Tensor Data Mover (tensor_load_to_lds + s_wait_tensorcnt).
__global__ void k_scan(const float* __restrict__ xs, const float* __restrict__ dts,
                       const float* __restrict__ xdbl, const float* __restrict__ A_logs,
                       const float* __restrict__ Ds, float* __restrict__ ys) {
  __shared__ float sBC[32 * SCH];   // row (0..31 = B rows then C rows) x col (0..SCH-1)
  int bk = blockIdx.x, k = bk & 3;
  int d = threadIdx.x;
  float Aa[NS], h[NS];
#pragma unroll
  for (int n = 0; n < NS; ++n) {
    Aa[n] = -__expf(A_logs[((size_t)(k * DI + d)) * NS + n]);
    h[n]  = 0.f;
  }
  float Dk = Ds[k * DI + d];
  const float* xsb  = xs  + ((size_t)(bk * DI + d)) * LL;
  const float* dtsb = dts + ((size_t)(bk * DI + d)) * LL;
  const float* xdb  = xdbl + (size_t)bk * 36 * LL;   // rows RR..RR+31 hold B then C
  float* yb = ys + ((size_t)(bk * DI + d)) * LL;

  // invariant parts of the tensor DMA descriptor
  unsigned lds_off = (unsigned)(unsigned long long)(const void*)&sBC[0];
  u32x8 g1;
  g1[0] = (2u << 16);            // workgroup_mask=0, data_size=4B
  g1[1] = (4096u & 0xFFFFu) << 16; // tensor_dim0 = 4096 (low16)
  g1[2] = (32u << 16);           // tensor_dim0 hi=0 | tensor_dim1 = 32
  g1[3] = ((unsigned)SCH) << 16; // tensor_dim1 hi=0 | tile_dim0 = SCH
  g1[4] = 32u;                   // tile_dim1 = 32, tile_dim2 = 0
  g1[5] = 4096u;                 // tensor_dim0_stride = 4096 (low32)
  g1[6] = 0u;
  g1[7] = 0u;

  for (int l0 = 0; l0 < LL; l0 += SCH) {
    __syncthreads();                       // previous chunk fully consumed
    if (threadIdx.x < 32) {                // wave 0 issues the TDM transfer
      unsigned long long ga = (unsigned long long)(const void*)(xdb + (size_t)RR * LL + l0);
      u32x4 g0;
      g0[0] = 1u;                                            // count=1 user descriptor
      g0[1] = lds_off;                                       // LDS byte address
      g0[2] = (unsigned)ga;                                  // global addr lo
      g0[3] = (unsigned)((ga >> 32) & 0x1FFFFFFu) | (2u << 30); // addr hi | type=2
      asm volatile("tensor_load_to_lds %0, %1" :: "s"(g0), "s"(g1) : "memory");
      __builtin_amdgcn_s_wait_tensorcnt(0);
    }
    if (l0 + SCH < LL) {
      __builtin_prefetch(dtsb + l0 + SCH, 0, 1);
      __builtin_prefetch(xsb + l0 + SCH, 0, 1);
    }
    __syncthreads();                       // chunk visible to all waves
#pragma unroll 4
    for (int lc = 0; lc < SCH; ++lc) {
      int l = l0 + lc;
      float dt = dtsb[l], u = xsb[l];
      float y = 0.f, dtu = dt * u;
#pragma unroll
      for (int n = 0; n < NS; ++n) {
        float e = __expf(dt * Aa[n]);
        h[n] = h[n] * e + dtu * sBC[n * SCH + lc];
        y += h[n] * sBC[(NS + n) * SCH + lc];
      }
      yb[l] = y + Dk * u;
    }
  }
}

// (8) merge 4 directions, out-LN over DI, gate with silu(z) -> y_act (B,L,DI)
__global__ void k_combine(const float* __restrict__ ys, const float* __restrict__ xz,
                          const float* __restrict__ g, const float* __restrict__ be,
                          float* __restrict__ yact) {
  int m = blockIdx.x * blockDim.x + threadIdx.x;
  if (m >= MTOT) return;
  int b = m >> 12, l = m & (LL - 1);
  int h = l >> 6, w = l & 63, lT = (w << 6) | h;
  const float* y0 = ys + ((size_t)(b * KD + 0)) * DI * LL;
  const float* y1 = ys + ((size_t)(b * KD + 1)) * DI * LL;
  const float* y2 = ys + ((size_t)(b * KD + 2)) * DI * LL;
  const float* y3 = ys + ((size_t)(b * KD + 3)) * DI * LL;
  float* o = yact + (size_t)m * DI;
  float s = 0.f, s2 = 0.f;
  for (int d = 0; d < DI; ++d) {
    float v = y0[(size_t)d * LL + l] + y1[(size_t)d * LL + lT]
            + y2[(size_t)d * LL + (LL - 1 - l)] + y3[(size_t)d * LL + (LL - 1 - lT)];
    o[d] = v; s += v; s2 += v * v;
  }
  float mu = s / DI, var = s2 / DI - mu * mu;
  float rs = rsqrtf(var + LN_EPS);
  const float* zrow = xz + (size_t)m * 256 + DI;
  for (int d = 0; d < DI; ++d)
    o[d] = ((o[d] - mu) * rs * g[d] + be[d]) * siluf(zrow[d]);
}

// (9) out_proj GEMM + residual(x NCHW): xh (B,L,C). one wave: 16x64 (all 4 N-tiles)
__global__ void k_outproj(const float* __restrict__ A, const float* __restrict__ W,
                          const float* __restrict__ bias, const float* __restrict__ xres,
                          float* __restrict__ xh) {
  int m0 = blockIdx.x * 16;
  v8f acc[4] = {};
#pragma unroll
  for (int kk = 0; kk < DI; kk += 32) {
    v16h a = frag_rowmajor(A, DI, m0, kk);
#pragma unroll
    for (int t = 0; t < 4; ++t) {
      v16h b = frag_rowmajor(W, DI, 16 * t, kk);
      acc[t] = WMMA_F16(a, b, acc[t]);
    }
  }
  int lane = threadIdx.x & 31, half = lane >> 4, r = lane & 15;
#pragma unroll
  for (int t = 0; t < 4; ++t) {
    int n = 16 * t + r;
    float bn = bias[n];
#pragma unroll
    for (int i = 0; i < 8; ++i) {
      int m = m0 + i + 8 * half, b2 = m >> 12, l = m & (LL - 1);
      xh[(size_t)m * CC + n] = acc[t][i] + bn + xres[((size_t)(b2 * CC + n)) * LL + l];
    }
  }
}

// (10) channel LayerNorm (over C) of xo=NCHW(xh) -> cl (B,C,L)
__global__ void k_cln(const float* __restrict__ xh, const float* __restrict__ g,
                      const float* __restrict__ be, float* __restrict__ cl) {
  int m = blockIdx.x * blockDim.x + threadIdx.x;
  if (m >= MTOT) return;
  int b = m >> 12, l = m & (LL - 1);
  const float* row = xh + (size_t)m * CC;
  float s = 0.f, s2 = 0.f;
  for (int c = 0; c < CC; ++c) { float v = row[c]; s += v; s2 += v * v; }
  float mu = s / CC, var = s2 / CC - mu * mu;
  float rs = rsqrtf(var + CN_EPS);
  for (int c = 0; c < CC; ++c)
    cl[((size_t)(b * CC + c)) * LL + l] = (row[c] - mu) * rs * g[c] + be[c];
}

// (11) SAFM multi-scale: maxpool(2^i) -> dwconv3 -> nearest upsample, per channel group
__global__ void k_safm_feat(const float* __restrict__ cl, const float* __restrict__ mw,
                            const float* __restrict__ mb, float* __restrict__ feat) {
  int t = blockIdx.x * blockDim.x + threadIdx.x;
  if (t >= BB * CC * LL) return;
  int l = t & (LL - 1), c = (t >> 12) & 63, b = t >> 18;
  int i = c >> 4, wc = c & 15, kp = 1 << i, Hp = HH >> i;
  int h = l >> 6, w = l & 63, ph = h >> i, pw = w >> i;
  const float* src = cl + ((size_t)(b * CC + c)) * LL;
  float acc = mb[i * 16 + wc];
  for (int dy = 0; dy < 3; ++dy) {
    int qy = ph + dy - 1; if (qy < 0 || qy >= Hp) continue;
    for (int dx = 0; dx < 3; ++dx) {
      int qx = pw + dx - 1; if (qx < 0 || qx >= Hp) continue;
      float pv = -INFINITY;
      for (int yy = qy * kp; yy < qy * kp + kp; ++yy)
        for (int xx = qx * kp; xx < qx * kp + kp; ++xx) {
          float v = src[yy * WW + xx];
          pv = v > pv ? v : pv;
        }
      acc += mw[((i * 16 + wc) * 3 + dy) * 3 + dx] * pv;
    }
  }
  feat[t] = acc;
}

// (12) aggr 1x1 GEMM (A gathered NCHW) + GELU gate + final residual -> d_out (NCHW)
// one wave: 16x64 (all 4 N-tiles), NCHW A fragment reused 4x
__global__ void k_aggr(const float* __restrict__ feat, const float* __restrict__ W,
                       const float* __restrict__ bias, const float* __restrict__ cl,
                       const float* __restrict__ xh, float* __restrict__ out) {
  int m0 = blockIdx.x * 16;
  v8f acc[4] = {};
#pragma unroll
  for (int kk = 0; kk < CC; kk += 32) {
    v16h a = frag_nchw(feat, CC, m0, kk);
#pragma unroll
    for (int t = 0; t < 4; ++t) {
      v16h b = frag_rowmajor(W, CC, 16 * t, kk);
      acc[t] = WMMA_F16(a, b, acc[t]);
    }
  }
  int lane = threadIdx.x & 31, half = lane >> 4, r = lane & 15;
#pragma unroll
  for (int t = 0; t < 4; ++t) {
    int n = 16 * t + r;
    float bn = bias[n];
#pragma unroll
    for (int i = 0; i < 8; ++i) {
      int m = m0 + i + 8 * half, b2 = m >> 12, l = m & (LL - 1);
      size_t nchw = ((size_t)(b2 * CC + n)) * LL + l;
      float gv = geluf(acc[t][i] + bn);
      out[nchw] = xh[(size_t)m * CC + n] + gv * cl[nchw];
    }
  }
}

// ================= launch =================
extern "C" void kernel_launch(void* const* d_in, const int* in_sizes, int n_in,
                              void* d_out, int out_size, void* d_ws, size_t ws_size,
                              hipStream_t stream) {
  const float* x        = (const float*)d_in[0];
  const float* ln1_g    = (const float*)d_in[1];
  const float* ln1_b    = (const float*)d_in[2];
  const float* inproj_w = (const float*)d_in[3];
  const float* inproj_b = (const float*)d_in[4];
  const float* conv_w   = (const float*)d_in[5];
  const float* conv_b   = (const float*)d_in[6];
  const float* xproj_w  = (const float*)d_in[7];
  const float* dtproj_w = (const float*)d_in[8];
  const float* dtproj_b = (const float*)d_in[9];
  const float* A_logs   = (const float*)d_in[10];
  const float* Ds       = (const float*)d_in[11];
  const float* onorm_g  = (const float*)d_in[12];
  const float* onorm_b  = (const float*)d_in[13];
  const float* oproj_w  = (const float*)d_in[14];
  const float* oproj_b  = (const float*)d_in[15];
  const float* mfr_w    = (const float*)d_in[16];
  const float* mfr_b    = (const float*)d_in[17];
  const float* aggr_w   = (const float*)d_in[18];
  const float* aggr_b   = (const float*)d_in[19];
  const float* cn_g     = (const float*)d_in[20];
  const float* cn_b     = (const float*)d_in[21];
  float* out = (float*)d_out;

  // workspace layout (floats)
  float* ws = (float*)d_ws;
  float* xn   = ws;                       // 2,097,152
  float* xz   = xn   + (size_t)2097152;   // 8,388,608
  float* xc   = xz   + (size_t)8388608;   // 4,194,304
  float* xs   = xc   + (size_t)4194304;   // 16,777,216
  float* xdbl = xs   + (size_t)16777216;  // 4,718,592
  float* dts  = xdbl + (size_t)4718592;   // 16,777,216
  float* ys   = dts  + (size_t)16777216;  // 16,777,216
  float* yact = ys   + (size_t)16777216;  // 4,194,304
  float* xh   = yact + (size_t)4194304;   // 2,097,152
  float* cl   = xh   + (size_t)2097152;   // 2,097,152
  float* feat = cl   + (size_t)2097152;   // 2,097,152

  k_ln1<<<dim3(MTOT / 256), dim3(256), 0, stream>>>(x, ln1_g, ln1_b, xn);
  k_inproj<<<dim3(4, MTOT / 16), dim3(32), 0, stream>>>(xn, inproj_w, inproj_b, xz);
  k_conv_silu<<<dim3((BB * DI * LL) / 256), dim3(256), 0, stream>>>(xz, conv_w, conv_b, xc);
  k_xs_build<<<dim3((BB * KD * DI * LL) / 256), dim3(256), 0, stream>>>(xc, xs);
  k_xdbl<<<dim3(LL / 16, BB * KD), dim3(32), 0, stream>>>(xproj_w, xs, xdbl);
  k_dt<<<dim3((BB * KD * DI * LL) / 256), dim3(256), 0, stream>>>(xdbl, dtproj_w, dtproj_b, dts);
  k_scan<<<dim3(BB * KD), dim3(DI), 0, stream>>>(xs, dts, xdbl, A_logs, Ds, ys);
  k_combine<<<dim3(MTOT / 128), dim3(128), 0, stream>>>(ys, xz, onorm_g, onorm_b, yact);
  k_outproj<<<dim3(MTOT / 16), dim3(32), 0, stream>>>(yact, oproj_w, oproj_b, x, xh);
  k_cln<<<dim3(MTOT / 256), dim3(256), 0, stream>>>(xh, cn_g, cn_b, cl);
  k_safm_feat<<<dim3((BB * CC * LL) / 256), dim3(256), 0, stream>>>(cl, mfr_w, mfr_b, feat);
  k_aggr<<<dim3(MTOT / 16), dim3(32), 0, stream>>>(feat, aggr_w, aggr_b, cl, xh, out);
}
